// Recurrence_47467978556129
// MI455X (gfx1250) — compile-verified, matrix operand order checked by reading, not run on hardware
//
#include <hip/hip_runtime.h>
#include <hip/hip_bf16.h>
#include <math.h>

typedef __attribute__((ext_vector_type(16))) _Float16 v16h;
typedef __attribute__((ext_vector_type(8)))  _Float16 v8h;
typedef __attribute__((ext_vector_type(8)))  float    v8f;

#define T_    64
#define N_    256
#define HID_  512
#define EE    128
#define CH_   64
#define NE_   4
#define NA_   16
#define NL_   16
#define OUTC  1082      // sum(SIZES)
#define XDIM  704       // CH + E + HID
#define G3H   1536      // 3*HID
#define R3E   384       // 3*E

// ---------------------------------------------------------------------------
// WMMA fragment helpers (16x16x32 f16, wave32).
// Per-lane K layout (A and B symmetric): lane group g = lane>>4 holds
// K = {8g .. 8g+7} and {16+8g .. 16+8g+7}  -> two contiguous 16B loads.
// C/D layout: element (m,n): vgpr = m&7, lane = (n&15) + 16*(m>>3).
// ---------------------------------------------------------------------------
__device__ __forceinline__ v16h frag_from(const _Float16* p) {
  union { v16h v; v8h h[2]; } u;
  u.h[0] = *(const v8h*)(p);
  u.h[1] = *(const v8h*)(p + 16);
  return u.v;
}

__device__ __forceinline__ float sigmf(float x) { return 1.f / (1.f + __expf(-x)); }

__device__ __forceinline__ void softmax_n(const float* in, float* out, int m) {
  float mx = in[0];
  for (int i = 1; i < m; ++i) mx = in[i] > mx ? in[i] : mx;
  float s = 0.f;
  for (int i = 0; i < m; ++i) { float e = __expf(in[i] - mx); out[i] = e; s += e; }
  float inv = 1.f / s;
  for (int i = 0; i < m; ++i) out[i] *= inv;
}

// ---------------------------------------------------------------------------
// f32 -> f16 weight conversion
// ---------------------------------------------------------------------------
__global__ void f32_to_f16_kernel(const float* __restrict__ src,
                                  _Float16* __restrict__ dst, int n) {
  int i = blockIdx.x * 256 + threadIdx.x;
  if (i < n) dst[i] = (_Float16)src[i];
}

// ---------------------------------------------------------------------------
// M[n,l,e] = mean_c et_w[lines[n,l,c] + c*VCOL, e]   (stored f16)
// ---------------------------------------------------------------------------
__global__ void embed_kernel(const float* __restrict__ etW,
                             const int* __restrict__ lines,
                             _Float16* __restrict__ Mh) {
  int idx = blockIdx.x * 256 + threadIdx.x;           // N*NL*E
  if (idx >= N_ * NL_ * EE) return;
  int e = idx & (EE - 1);
  int l = (idx >> 7) & (NL_ - 1);
  int n = idx >> 11;
  float s = 0.f;
  #pragma unroll
  for (int c = 0; c < 4; ++c) {
    int r = lines[(n * NL_ + l) * 4 + c] + c * 8;
    s += etW[(size_t)r * EE + e];
  }
  Mh[idx] = (_Float16)(s * 0.25f);
}

// ---------------------------------------------------------------------------
// Fused 16-step GRU over 16 batch rows per block (4 waves, 24 col tiles of
// 3E=384). Input rows are gathered on the fly from M (rolled view), hidden
// state lives in LDS, all GEMMs are v_wmma_f32_16x16x32_f16.
// ---------------------------------------------------------------------------
__global__ __launch_bounds__(128) void gru_seq_kernel(
    const _Float16* __restrict__ Mh,                   // (N, NL, E)
    const _Float16* __restrict__ Wih,                  // (384,128) f16
    const _Float16* __restrict__ Whh,                  // (384,128) f16
    const float* __restrict__ bih, const float* __restrict__ bhh,
    _Float16* __restrict__ G, int dir)                 // (4096, NL, E) f16
{
  __shared__ __align__(16) _Float16 hf16[16 * EE];
  __shared__ float h32[16 * EE];
  __shared__ float gi_s[16 * R3E];
  __shared__ float gh_s[16 * R3E];

  const int tid  = threadIdx.x;
  const int wave = tid >> 5;
  const int l    = tid & 31;
  const int m    = l & 15;
  const int g    = l >> 4;

  for (int i = tid; i < 16 * EE; i += 128) { h32[i] = 0.f; hf16[i] = (_Float16)0.f; }
  __syncthreads();

  const int r    = blockIdx.x * 16 + m;   // batch row for this lane's A-frag
  const int n    = r & (N_ - 1);
  const int lrow = r >> 8;

  for (int t = 0; t < NL_; ++t) {
    const int pos  = dir ? (NL_ - 1 - t) : t;
    const int midx = (lrow + pos) & (NL_ - 1);
    const _Float16* xrow = Mh + ((size_t)(n * NL_ + midx)) * EE + 8 * g;
    const _Float16* hrow = hf16 + m * EE + 8 * g;

    v16h ax[4], ah[4];
    #pragma unroll
    for (int ks = 0; ks < 4; ++ks) {
      ax[ks] = frag_from(xrow + ks * 32);
      ah[ks] = frag_from(hrow + ks * 32);
    }

    #pragma unroll
    for (int i = 0; i < 6; ++i) {
      const int ct = wave * 6 + i;
      v8f acc = {};
      #pragma unroll
      for (int ks = 0; ks < 4; ++ks) {
        v16h b = frag_from(Wih + (size_t)(ct * 16 + m) * EE + ks * 32 + 8 * g);
        acc = __builtin_amdgcn_wmma_f32_16x16x32_f16(false, ax[ks], false, b,
                                                     (short)0, acc, false, false);
      }
      #pragma unroll
      for (int q = 0; q < 8; ++q) gi_s[(g * 8 + q) * R3E + ct * 16 + m] = acc[q];

      v8f acch = {};
      #pragma unroll
      for (int ks = 0; ks < 4; ++ks) {
        v16h b = frag_from(Whh + (size_t)(ct * 16 + m) * EE + ks * 32 + 8 * g);
        acch = __builtin_amdgcn_wmma_f32_16x16x32_f16(false, ah[ks], false, b,
                                                      (short)0, acch, false, false);
      }
      #pragma unroll
      for (int q = 0; q < 8; ++q) gh_s[(g * 8 + q) * R3E + ct * 16 + m] = acch[q];
    }
    __syncthreads();

    for (int idx = tid; idx < 16 * EE; idx += 128) {
      const int row = idx >> 7;
      const int e   = idx & (EE - 1);
      float ir  = gi_s[row * R3E + e]          + bih[e];
      float iz  = gi_s[row * R3E + EE + e]     + bih[EE + e];
      float in2 = gi_s[row * R3E + 2 * EE + e] + bih[2 * EE + e];
      float hr  = gh_s[row * R3E + e]          + bhh[e];
      float hz  = gh_s[row * R3E + EE + e]     + bhh[EE + e];
      float hn  = gh_s[row * R3E + 2 * EE + e] + bhh[2 * EE + e];
      float rg = sigmf(ir + hr);
      float zg = sigmf(iz + hz);
      float ng = tanhf(in2 + rg * hn);
      float hv = (1.f - zg) * ng + zg * h32[idx];
      h32[idx]  = hv;
      hf16[idx] = (_Float16)hv;
      const int rr = blockIdx.x * 16 + row;
      G[((size_t)rr * NL_ + pos) * EE + e] = (_Float16)hv;
    }
    __syncthreads();
  }
}

// ---------------------------------------------------------------------------
// Generic WMMA GEMM: Y(rows x Cout) = X(rows x K) @ W(Cout x K)^T (+bias,relu)
// grid = (rows/16, Cout/64), block = 128 (4 waves, one 16x16 tile each).
// ---------------------------------------------------------------------------
__global__ __launch_bounds__(128) void gemm_wmma_kernel(
    const _Float16* __restrict__ X, const _Float16* __restrict__ W,
    const float* __restrict__ bias, float* __restrict__ Y,
    _Float16* __restrict__ Yh, int K, int Cout, int relu)
{
  const int wave = threadIdx.x >> 5;
  const int l    = threadIdx.x & 31;
  const int m    = l & 15;
  const int g    = l >> 4;
  const int row0 = blockIdx.x * 16;
  const int col0 = (blockIdx.y * 4 + wave) * 16;
  if (col0 >= Cout) return;

  const _Float16* xrow = X + (size_t)(row0 + m) * K + 8 * g;
  const _Float16* wrow = W + (size_t)(col0 + m) * K + 8 * g;

  v8f acc = {};
  for (int k0 = 0; k0 < K; k0 += 32) {
    __builtin_prefetch(wrow + k0 + 256, 0, 1);          // global_prefetch_b8
    v16h a = frag_from(xrow + k0);
    v16h b = frag_from(wrow + k0);
    acc = __builtin_amdgcn_wmma_f32_16x16x32_f16(false, a, false, b,
                                                 (short)0, acc, false, false);
  }
  const float bv = bias ? bias[col0 + m] : 0.f;
  #pragma unroll
  for (int q = 0; q < 8; ++q) {
    float v = acc[q] + bv;
    if (relu) v = v > 0.f ? v : 0.f;
    const size_t off = (size_t)(row0 + g * 8 + q) * Cout + col0 + m;
    if (Y)  Y[off]  = v;
    if (Yh) Yh[off] = (_Float16)v;
  }
}

// ---------------------------------------------------------------------------
// B = sigmoid(Bh @ b_W2^T + b_b2): rows=65536, out=4   (VALU head)
// ---------------------------------------------------------------------------
__global__ void bsig_kernel(const _Float16* __restrict__ Bh,
                            const float* __restrict__ bW2,
                            const float* __restrict__ bb2,
                            float* __restrict__ Bout) {
  int idx = blockIdx.x * 256 + threadIdx.x;   // 65536*4
  if (idx >= 65536 * 4) return;
  int o = idx & 3, row = idx >> 2;
  const _Float16* p = Bh + (size_t)row * EE;
  float s = bb2[o];
  for (int e = 0; e < EE; ++e) s += (float)p[e] * bW2[o * EE + e];
  Bout[idx] = sigmf(s);
}

// ---------------------------------------------------------------------------
// PP: flip/stack/cumprod chain. One thread per (l, n, e): 32-step serial scan.
// ---------------------------------------------------------------------------
__global__ void pp_kernel(const float* __restrict__ B0,
                          const float* __restrict__ B1,
                          float* __restrict__ PP) {
  int idx = blockIdx.x * 256 + threadIdx.x;   // 16*256*4
  if (idx >= NL_ * N_ * NE_) return;
  int e  = idx & 3;
  int n  = (idx >> 2) & (N_ - 1);
  int ll = idx >> 10;
  const size_t rowbase = ((size_t)ll * N_ + n) * NL_;

  float B2[32];
  #pragma unroll
  for (int jj = 0; jj < NL_; ++jj) {
    B2[2 * jj]     = B0[(rowbase + jj) * 4 + e];
    B2[2 * jj + 1] = B1[(rowbase + (NL_ - 1 - jj)) * 4 + e];
  }
  // zl[j] = (j==0 || j==31) ? 0 : B2[j];   B4[j] = (j==31) ? 1 : zl[j]
  // C[j]  = prod_{i<=j} (1 - zl[(i-1) mod 32])
  float PPv[32];
  float prod = 1.f;
  #pragma unroll
  for (int j = 0; j < 32; ++j) {
    float rz = (j == 0) ? 0.f : ((j - 1 == 0) ? 0.f : B2[j - 1]);
    prod *= (1.f - rz);
    float b4 = (j == 31) ? 1.f : ((j == 0) ? 0.f : B2[j]);
    PPv[j] = b4 * prod;
  }
  // PP[k]: k<16 -> b2[15-k] = PPv[2*(15-k)+1]; k>=16 -> f2[k-16] = PPv[2*(k-16)]
  #pragma unroll
  for (int k = 0; k < 32; ++k) {
    float v = (k < NL_) ? PPv[2 * (NL_ - 1 - k) + 1] : PPv[2 * (k - NL_)];
    PP[(((size_t)ll * N_ + n) * 32 + k) * 4 + e] = v;
  }
}

// ---------------------------------------------------------------------------
// Initial recurrent state from rnn_hxs
// ---------------------------------------------------------------------------
__global__ __launch_bounds__(256) void init_state_kernel(
    const float* __restrict__ rnn, float* __restrict__ h32,
    _Float16* __restrict__ h16, float* __restrict__ h2_32,
    _Float16* __restrict__ h2_16, int* __restrict__ ptr,
    int* __restrict__ aprev0) {
  const int n = blockIdx.x, tid = threadIdx.x;
  __shared__ int nz;
  if (tid == 0) nz = 0;
  __syncthreads();
  const float* row = rnn + (size_t)n * OUTC;
  int local = 0;
  for (int i = tid; i < OUTC; i += 256) if (row[i] != 0.f) local = 1;
  if (local) nz = 1;                                  // benign race (all write 1)
  __syncthreads();
  for (int i = tid; i < HID_; i += 256) {
    float hv  = row[6 + i];
    float h2v = row[518 + i];
    h32[(size_t)n * HID_ + i]   = hv;  h16[(size_t)n * HID_ + i]   = (_Float16)hv;
    h2_32[(size_t)n * HID_ + i] = h2v; h2_16[(size_t)n * HID_ + i] = (_Float16)h2v;
  }
  if (tid == 0) {
    ptr[n]    = (int)row[4];
    aprev0[n] = nz ? (int)row[0] : (NA_ - 1);
  }
}

// ---------------------------------------------------------------------------
// Per-step: 1x1 conv + spatial max + build x = [o | M[n,ptr] | ea_w[a_prev]]
// ---------------------------------------------------------------------------
__global__ __launch_bounds__(256) void conv_x_kernel(
    const float* __restrict__ obs, const float* __restrict__ cvW,
    const float* __restrict__ cvb, const _Float16* __restrict__ Mh,
    const float* __restrict__ eaW, const int* __restrict__ actions,
    const int* __restrict__ aprev0, const int* __restrict__ ptr,
    _Float16* __restrict__ xh, int t) {
  const int n = blockIdx.x, tid = threadIdx.x;
  __shared__ float o_s[CH_];
  if (tid < CH_) {
    const float* ob = obs + ((size_t)t * N_ + n) * 16 * 64;
    float mx = 0.f;                                   // relu outputs are >= 0
    for (int hw = 0; hw < 64; ++hw) {
      float s = cvb[tid];
      #pragma unroll
      for (int c = 0; c < 16; ++c) s += ob[c * 64 + hw] * cvW[tid * 16 + c];
      s = s > 0.f ? s : 0.f;
      mx = s > mx ? s : mx;
    }
    o_s[tid] = mx;
  }
  __syncthreads();
  const int ap = (t == 0) ? aprev0[n] : actions[((size_t)(t - 1) * N_ + n) * 4 + 0];
  const int pt = ptr[n];
  for (int j = tid; j < XDIM; j += 256) {
    float v;
    if (j < CH_)            v = o_s[j];
    else if (j < CH_ + EE)  v = (float)Mh[((size_t)n * NL_ + pt) * EE + (j - CH_)];
    else                    v = eaW[(size_t)ap * HID_ + (j - CH_ - EE)];
    xh[(size_t)n * XDIM + j] = (_Float16)v;
  }
}

// ---------------------------------------------------------------------------
// GRU gate math for the main scan (HID=512, gi/gh strides 1536)
// ---------------------------------------------------------------------------
__global__ __launch_bounds__(256) void gru_gate_kernel(
    const float* __restrict__ gi, const float* __restrict__ gh,
    const float* __restrict__ bih, const float* __restrict__ bhh,
    const float* __restrict__ h_old, float* __restrict__ h_out,
    _Float16* __restrict__ h_out16) {
  int idx = blockIdx.x * 256 + threadIdx.x;
  if (idx >= N_ * HID_) return;
  int nn = idx >> 9, e = idx & (HID_ - 1);
  size_t b = (size_t)nn * G3H;
  float ir  = gi[b + e]            + bih[e];
  float iz  = gi[b + HID_ + e]     + bih[HID_ + e];
  float in2 = gi[b + 2 * HID_ + e] + bih[2 * HID_ + e];
  float hr  = gh[b + e]            + bhh[e];
  float hz  = gh[b + HID_ + e]     + bhh[HID_ + e];
  float hn  = gh[b + 2 * HID_ + e] + bhh[2 * HID_ + e];
  float rg = sigmf(ir + hr);
  float zg = sigmf(iz + hz);
  float ng = tanhf(in2 + rg * hn);
  float hv = (1.f - zg) * ng + zg * h_old[idx];
  h_out[idx] = hv;
  h_out16[idx] = (_Float16)hv;
}

// ---------------------------------------------------------------------------
// Heads + pointer update + h2 mixing + output row writeback
// ---------------------------------------------------------------------------
__global__ __launch_bounds__(128) void head_kernel(
    const float* __restrict__ z1, const float* __restrict__ z2,
    const float* __restrict__ h_new, const float* __restrict__ h2n,
    float* __restrict__ h2_32, _Float16* __restrict__ h2_16,
    const float* __restrict__ PP, int* __restrict__ ptr,
    const int* __restrict__ aprev0, const int* __restrict__ actions,
    const float* __restrict__ uW, const float* __restrict__ ub,
    const float* __restrict__ aW, const float* __restrict__ ab,
    const float* __restrict__ cW, const float* __restrict__ cb,
    const float* __restrict__ dgW, const float* __restrict__ dgb,
    const float* __restrict__ agW, const float* __restrict__ agb,
    float* __restrict__ out, int t) {
  const int n = blockIdx.x, tid = threadIdx.x;
  __shared__ float dots[25];
  __shared__ float sm_dg[2], sm_ag[2], sm_u[4], sm_a[16];
  __shared__ int s_ptr;
  if (tid == 0) s_ptr = ptr[n];

  const float* z1n = z1 + (size_t)n * HID_;
  const float* z2n = z2 + (size_t)n * HID_;
  if (tid < 25) {
    const float *wsrc, *zsrc; float bb;
    if (tid < 2)       { zsrc = z1n; wsrc = dgW + tid * HID_;       bb = dgb[tid]; }
    else if (tid < 4)  { zsrc = z1n; wsrc = agW + (tid - 2) * HID_; bb = agb[tid - 2]; }
    else if (tid < 8)  { zsrc = z2n; wsrc = uW  + (tid - 4) * HID_; bb = ub[tid - 4]; }
    else if (tid < 24) { zsrc = z2n; wsrc = aW  + (tid - 8) * HID_; bb = ab[tid - 8]; }
    else               { zsrc = z2n; wsrc = cW;                     bb = cb[0]; }
    float s = bb;
    for (int e = 0; e < HID_; ++e) s += zsrc[e] * wsrc[e];
    dots[tid] = s;
  }
  __syncthreads();
  if (tid == 0) {
    softmax_n(dots + 0, sm_dg, 2);
    softmax_n(dots + 2, sm_ag, 2);
    softmax_n(dots + 4, sm_u, 4);
    softmax_n(dots + 8, sm_a, 16);
  }
  __syncthreads();

  const int a_t  = actions[((size_t)t * N_ + n) * 4 + 0];
  const int d_t  = actions[((size_t)t * N_ + n) * 4 + 1];
  const int ag_t = actions[((size_t)t * N_ + n) * 4 + 2];
  const int dg_t = actions[((size_t)t * N_ + n) * 4 + 3];
  const int ap   = (t == 0) ? aprev0[n] : actions[((size_t)(t - 1) * N_ + n) * 4 + 0];
  const float fdg = (float)dg_t, fag = (float)ag_t;
  const size_t base = ((size_t)t * N_ + n) * OUTC;

  if (tid < 32) {
    const float* pp = PP + (((size_t)s_ptr * N_ + n) * 32 + tid) * 4;
    float draw = 0.f;
    #pragma unroll
    for (int e = 0; e < 4; ++e) draw += pp[e] * sm_u[e];
    float dold = (tid == NL_) ? 1.f : 0.f;
    out[base + 1046 + tid] = fdg * draw + (1.f - fdg) * dold;
  }
  if (tid < 16) {
    float aold = (tid == ap) ? 1.f : 0.f;
    out[base + 1030 + tid] = fag * sm_a[tid] + (1.f - fag) * aold;
  }
  if (tid < 2) {
    out[base + 1078 + tid] = sm_ag[tid];
    out[base + 1080 + tid] = sm_dg[tid];
  }
  int ptr_new = s_ptr + d_t - NL_;
  ptr_new = ptr_new < 0 ? 0 : (ptr_new > NL_ - 1 ? NL_ - 1 : ptr_new);
  if (tid == 0) {
    out[base + 0] = (float)a_t;
    out[base + 1] = (float)d_t;
    out[base + 2] = (float)ag_t;
    out[base + 3] = (float)dg_t;
    out[base + 4] = (float)ptr_new;
    out[base + 5] = dots[24];
    ptr[n] = ptr_new;
  }
  for (int e = tid; e < HID_; e += 128) {
    const size_t hoff = (size_t)n * HID_ + e;
    out[base + 6 + e] = h_new[hoff];
    float h2v = fdg * h2n[hoff] + (1.f - fdg) * h2_32[hoff];
    h2_32[hoff] = h2v;
    h2_16[hoff] = (_Float16)h2v;
    out[base + 518 + e] = h2v;
  }
}

// ---------------------------------------------------------------------------
extern "C" void kernel_launch(void* const* d_in, const int* in_sizes, int n_in,
                              void* d_out, int out_size, void* d_ws, size_t ws_size,
                              hipStream_t stream) {
  (void)in_sizes; (void)n_in; (void)out_size; (void)ws_size;
  // inputs
  const float* obs    = (const float*)d_in[0];
  const int*   lines  = (const int*)d_in[1];
  const int*   actions= (const int*)d_in[2];
  const float* rnn    = (const float*)d_in[3];
  const float* et_w   = (const float*)d_in[4];
  const float* ea_w   = (const float*)d_in[5];
  const float* Wih_f  = (const float*)d_in[6];
  const float* Whh_f  = (const float*)d_in[7];
  const float* bih_f  = (const float*)d_in[8];
  const float* bhh_f  = (const float*)d_in[9];
  const float* Wih_b  = (const float*)d_in[10];
  const float* Whh_b  = (const float*)d_in[11];
  const float* bih_b  = (const float*)d_in[12];
  const float* bhh_b  = (const float*)d_in[13];
  const float* g_Wih  = (const float*)d_in[14];
  const float* g_Whh  = (const float*)d_in[15];
  const float* g_bih  = (const float*)d_in[16];
  const float* g_bhh  = (const float*)d_in[17];
  const float* z_W1   = (const float*)d_in[18];
  const float* z_b1   = (const float*)d_in[19];
  const float* z_W2   = (const float*)d_in[20];
  const float* z_b2   = (const float*)d_in[21];
  const float* u_W    = (const float*)d_in[22];
  const float* u_b    = (const float*)d_in[23];
  const float* b_W1   = (const float*)d_in[24];
  const float* b_b1   = (const float*)d_in[25];
  const float* b_W2   = (const float*)d_in[26];
  const float* b_b2   = (const float*)d_in[27];
  const float* c_W    = (const float*)d_in[28];
  const float* c_b    = (const float*)d_in[29];
  const float* a_W    = (const float*)d_in[30];
  const float* a_b    = (const float*)d_in[31];
  const float* dg_W   = (const float*)d_in[32];
  const float* dg_b   = (const float*)d_in[33];
  const float* ag_W   = (const float*)d_in[34];
  const float* ag_b   = (const float*)d_in[35];
  const float* cv_W   = (const float*)d_in[36];
  const float* cv_b   = (const float*)d_in[37];
  float* out = (float*)d_out;

  // workspace carve-out (256B aligned)
  char* wp = (char*)d_ws;
  auto alloc = [&](size_t bytes) -> void* {
    void* p = (void*)wp;
    wp += (bytes + 255) & ~(size_t)255;
    return p;
  };
  _Float16* Mh    = (_Float16*)alloc((size_t)N_ * NL_ * EE * 2);
  _Float16* G0    = (_Float16*)alloc((size_t)NL_ * N_ * NL_ * EE * 2);
  _Float16* G1    = (_Float16*)alloc((size_t)NL_ * N_ * NL_ * EE * 2);
  _Float16* Bh16  = (_Float16*)alloc((size_t)65536 * EE * 2);
  float*    Bb0   = (float*)alloc((size_t)65536 * 4 * 4);
  float*    Bb1   = (float*)alloc((size_t)65536 * 4 * 4);
  float*    PPb   = (float*)alloc((size_t)NL_ * N_ * 32 * NE_ * 4);
  _Float16* WihFh = (_Float16*)alloc((size_t)R3E * EE * 2);
  _Float16* WhhFh = (_Float16*)alloc((size_t)R3E * EE * 2);
  _Float16* WihBh = (_Float16*)alloc((size_t)R3E * EE * 2);
  _Float16* WhhBh = (_Float16*)alloc((size_t)R3E * EE * 2);
  _Float16* gWihh = (_Float16*)alloc((size_t)G3H * XDIM * 2);
  _Float16* gWhhh = (_Float16*)alloc((size_t)G3H * HID_ * 2);
  _Float16* zW1h  = (_Float16*)alloc((size_t)HID_ * HID_ * 2);
  _Float16* zW2h  = (_Float16*)alloc((size_t)HID_ * HID_ * 2);
  _Float16* bW1h  = (_Float16*)alloc((size_t)EE * EE * 2);
  float*    h32   = (float*)alloc((size_t)N_ * HID_ * 4);
  _Float16* h16   = (_Float16*)alloc((size_t)N_ * HID_ * 2);
  float*    h2_32 = (float*)alloc((size_t)N_ * HID_ * 4);
  _Float16* h2_16 = (_Float16*)alloc((size_t)N_ * HID_ * 2);
  float*    h2n32 = (float*)alloc((size_t)N_ * HID_ * 4);
  _Float16* h2n16 = (_Float16*)alloc((size_t)N_ * HID_ * 2);
  _Float16* xh    = (_Float16*)alloc((size_t)N_ * XDIM * 2);
  float*    gi    = (float*)alloc((size_t)N_ * G3H * 4);
  float*    gh1   = (float*)alloc((size_t)N_ * G3H * 4);
  float*    gh2   = (float*)alloc((size_t)N_ * G3H * 4);
  _Float16* zt16  = (_Float16*)alloc((size_t)N_ * HID_ * 2);
  float*    z1b   = (float*)alloc((size_t)N_ * HID_ * 4);
  float*    z2b   = (float*)alloc((size_t)N_ * HID_ * 4);
  int*      ptrb  = (int*)alloc((size_t)N_ * 4);
  int*      ap0   = (int*)alloc((size_t)N_ * 4);

  auto cvt = [&](const float* s, _Float16* d, int cnt) {
    f32_to_f16_kernel<<<(cnt + 255) / 256, 256, 0, stream>>>(s, d, cnt);
  };
  cvt(Wih_f, WihFh, R3E * EE);   cvt(Whh_f, WhhFh, R3E * EE);
  cvt(Wih_b, WihBh, R3E * EE);   cvt(Whh_b, WhhBh, R3E * EE);
  cvt(g_Wih, gWihh, G3H * XDIM); cvt(g_Whh, gWhhh, G3H * HID_);
  cvt(z_W1,  zW1h,  HID_ * HID_); cvt(z_W2, zW2h, HID_ * HID_);
  cvt(b_W1,  bW1h,  EE * EE);

  // embedding M (f16)
  embed_kernel<<<(N_ * NL_ * EE + 255) / 256, 256, 0, stream>>>(et_w, lines, Mh);

  // bidirectional fused GRU (WMMA)
  gru_seq_kernel<<<dim3(NL_ * N_ / 16), 128, 0, stream>>>(
      Mh, WihFh, WhhFh, bih_f, bhh_f, G0, 0);
  gru_seq_kernel<<<dim3(NL_ * N_ / 16), 128, 0, stream>>>(
      Mh, WihBh, WhhBh, bih_b, bhh_b, G1, 1);

  // boundary MLP: Bh = relu(G @ b_W1^T + b_b1)  (WMMA), then sigmoid head
  gemm_wmma_kernel<<<dim3(65536 / 16, 2), 128, 0, stream>>>(
      G0, bW1h, b_b1, nullptr, Bh16, EE, EE, 1);
  bsig_kernel<<<(65536 * 4 + 255) / 256, 256, 0, stream>>>(Bh16, b_W2, b_b2, Bb0);
  gemm_wmma_kernel<<<dim3(65536 / 16, 2), 128, 0, stream>>>(
      G1, bW1h, b_b1, nullptr, Bh16, EE, EE, 1);
  bsig_kernel<<<(65536 * 4 + 255) / 256, 256, 0, stream>>>(Bh16, b_W2, b_b2, Bb1);

  // cumprod chain -> PP
  pp_kernel<<<(NL_ * N_ * NE_ + 255) / 256, 256, 0, stream>>>(Bb0, Bb1, PPb);

  // recurrent state init
  init_state_kernel<<<N_, 256, 0, stream>>>(rnn, h32, h16, h2_32, h2_16, ptrb, ap0);

  // -------- main scan: 64 sequential steps --------
  for (int t = 0; t < T_; ++t) {
    conv_x_kernel<<<N_, 256, 0, stream>>>(obs, cv_W, cv_b, Mh, ea_w, actions,
                                          ap0, ptrb, xh, t);
    // gi = x @ g_Wih^T (shared by h and h2 branches); gh per branch
    gemm_wmma_kernel<<<dim3(N_ / 16, G3H / 64), 128, 0, stream>>>(
        xh, gWihh, nullptr, gi, nullptr, XDIM, G3H, 0);
    gemm_wmma_kernel<<<dim3(N_ / 16, G3H / 64), 128, 0, stream>>>(
        h16, gWhhh, nullptr, gh1, nullptr, HID_, G3H, 0);
    gemm_wmma_kernel<<<dim3(N_ / 16, G3H / 64), 128, 0, stream>>>(
        h2_16, gWhhh, nullptr, gh2, nullptr, HID_, G3H, 0);
    gru_gate_kernel<<<(N_ * HID_ + 255) / 256, 256, 0, stream>>>(
        gi, gh1, g_bih, g_bhh, h32, h32, h16);
    gru_gate_kernel<<<(N_ * HID_ + 255) / 256, 256, 0, stream>>>(
        gi, gh2, g_bih, g_bhh, h2_32, h2n32, h2n16);
    // zeta(h_new) -> z1 ; zeta(h2n) -> z2  (2-layer relu MLP, WMMA)
    gemm_wmma_kernel<<<dim3(N_ / 16, HID_ / 64), 128, 0, stream>>>(
        h16, zW1h, z_b1, nullptr, zt16, HID_, HID_, 1);
    gemm_wmma_kernel<<<dim3(N_ / 16, HID_ / 64), 128, 0, stream>>>(
        zt16, zW2h, z_b2, z1b, nullptr, HID_, HID_, 1);
    gemm_wmma_kernel<<<dim3(N_ / 16, HID_ / 64), 128, 0, stream>>>(
        h2n16, zW1h, z_b1, nullptr, zt16, HID_, HID_, 1);
    gemm_wmma_kernel<<<dim3(N_ / 16, HID_ / 64), 128, 0, stream>>>(
        zt16, zW2h, z_b2, z2b, nullptr, HID_, HID_, 1);
    head_kernel<<<N_, 128, 0, stream>>>(
        z1b, z2b, h32, h2n32, h2_32, h2_16, PPb, ptrb, ap0, actions,
        u_W, u_b, a_W, a_b, c_W, c_b, dg_W, dg_b, ag_W, ag_b, out, t);
  }

  // second output = hx[-1:]
  hipMemcpyAsync(out + (size_t)T_ * N_ * OUTC,
                 out + (size_t)(T_ - 1) * N_ * OUTC,
                 (size_t)N_ * OUTC * sizeof(float),
                 hipMemcpyDeviceToDevice, stream);
}
